// EfficientAttention_42073499632181
// MI455X (gfx1250) — matrix operand order repaired and checked by reading, not compiled, not run on hardware
//
#include <hip/hip_runtime.h>
#include <hip/hip_bf16.h>
#include <stdint.h>

// ---------------- problem constants ----------------
#define B_SZ    4
#define S_SEQ   2048
#define H_DIM   1024
#define NHEADS  8
#define HD_DIM  128

typedef __attribute__((ext_vector_type(16))) __bf16 v16bf;
typedef __attribute__((ext_vector_type(8)))  float  v8f;
typedef __attribute__((ext_vector_type(4)))  __bf16 bf16x4;
typedef __attribute__((ext_vector_type(4)))  int    v4i;

union FragBF {
    v16bf    v;
    uint4    q[2];
    uint32_t u[8];
};

__device__ __forceinline__ v8f zero8() {
    v8f z = {0.f,0.f,0.f,0.f,0.f,0.f,0.f,0.f};
    return z;
}

__device__ __forceinline__ v8f wmma_bf16(const FragBF& a, const FragBF& b, v8f c) {
    return __builtin_amdgcn_wmma_f32_16x16x32_bf16(
        /*neg_a=*/false, a.v, /*neg_b=*/false, b.v,
        /*c_mod=*/(short)0, c, /*reuse_a=*/false, /*reuse_b=*/false);
}

// ---- CDNA5 async global->LDS copy (ASYNCcnt path), guarded with fallback ----
#if defined(__has_builtin)
#  if __has_builtin(__builtin_amdgcn_global_load_async_to_lds_b128) && \
      __has_builtin(__builtin_amdgcn_s_wait_asynccnt)
#    define HAVE_ASYNC_LDS 1
#  endif
#endif
#ifndef HAVE_ASYNC_LDS
#  define HAVE_ASYNC_LDS 0
#endif

__device__ __forceinline__ void async_copy16(const __bf16* g, __bf16* l) {
#if HAVE_ASYNC_LDS
    __builtin_amdgcn_global_load_async_to_lds_b128(
        (__attribute__((address_space(1))) v4i*)g,
        (__attribute__((address_space(3))) v4i*)l, 0, 0);
#else
    *(uint4*)l = *(const uint4*)g;
#endif
}
__device__ __forceinline__ void async_join() {
#if HAVE_ASYNC_LDS
    __builtin_amdgcn_s_wait_asynccnt(0);
#endif
}

// ---------------- fp32 -> bf16 conversion (vectorized) ----------------
__global__ void cvt_f32_to_bf16_v4(const float* __restrict__ in,
                                   __bf16* __restrict__ out, int n4) {
    int i = blockIdx.x * blockDim.x + threadIdx.x;
    if (i >= n4) return;
    float4 f = ((const float4*)in)[i];
    bf16x4 h;
    h[0] = (__bf16)f.x; h[1] = (__bf16)f.y;
    h[2] = (__bf16)f.z; h[3] = (__bf16)f.w;
    ((bf16x4*)out)[i] = h;
}

// ---------------- GEMM: C[m,n] = sum_k A[m,k] * W[n,k]  (X @ W^T) ----------------
// Block: 256 threads = 8 waves. Wave computes a 32(M) x 64(N) tile (8 WMMA / k-step).
// Grid: (M/32, N/512).
// MODE 0: bf16 out, scatter to [B,NH,S,HD]   (Q, K)
// MODE 1: bf16 out, scatter to [B,NH,HD,S]   (V transposed)
// MODE 2: fp32 out, row-major [M,N]          (final projection)
template <int MODE>
__global__ __launch_bounds__(256) void gemm_xwt(const __bf16* __restrict__ A,
                                                const __bf16* __restrict__ W,
                                                void* __restrict__ Cout,
                                                int M, int N, int Kdim) {
    const int lane   = threadIdx.x & 31;
    const int wave   = threadIdx.x >> 5;
    const int m0     = blockIdx.x * 32;
    const int n0     = blockIdx.y * 512 + wave * 64;
    const int col    = lane & 15;
    const int aHalfB = (lane & 16) ? 16 : 0;   // A-frag lane-half byte offset
    const int bHalfB = (lane & 16) ? 32 : 0;   // unused; B-frag also uses aHalfB

    const char* arow0 = (const char*)(A + (size_t)(m0 + col) * Kdim);
    const char* arow1 = arow0 + (size_t)16 * Kdim * 2;

    v8f acc[2][4];
#pragma unroll
    for (int i = 0; i < 2; ++i)
#pragma unroll
        for (int t = 0; t < 4; ++t) acc[i][t] = zero8();

    for (int k0 = 0; k0 < Kdim; k0 += 32) {
        FragBF af0, af1;
        // A-frag: segments at bytes 2*k0 + {aHalfB, 32+aHalfB}
        af0.q[0] = *(const uint4*)(arow0 + 2 * k0 + aHalfB);
        af0.q[1] = *(const uint4*)(arow0 + 2 * k0 + 32 + aHalfB);
        af1.q[0] = *(const uint4*)(arow1 + 2 * k0 + aHalfB);
        af1.q[1] = *(const uint4*)(arow1 + 2 * k0 + 32 + aHalfB);
#pragma unroll
        for (int t = 0; t < 4; ++t) {
            const char* wrow = (const char*)(W + (size_t)(n0 + t * 16 + col) * Kdim) + 2 * k0;
            FragBF bfg;
            bfg.q[0] = *(const uint4*)(wrow + aHalfB);
            bfg.q[1] = *(const uint4*)(wrow + 32 + aHalfB);
            acc[0][t] = wmma_bf16(af0, bfg, acc[0][t]);
            acc[1][t] = wmma_bf16(af1, bfg, acc[1][t]);
        }
    }

    const int mhalf = (lane & 16) ? 8 : 0;
#pragma unroll
    for (int i = 0; i < 2; ++i) {
#pragma unroll
        for (int t = 0; t < 4; ++t) {
#pragma unroll
            for (int r = 0; r < 8; ++r) {
                int m = m0 + i * 16 + r + mhalf;
                int n = n0 + t * 16 + col;
                float val = acc[i][t][r];
                if (MODE == 2) {
                    ((float*)Cout)[(size_t)m * N + n] = val;
                } else {
                    int b = m / S_SEQ, s = m % S_SEQ;
                    int h = n / HD_DIM, d = n % HD_DIM;
                    size_t idx;
                    if (MODE == 0)
                        idx = (((size_t)(b * NHEADS + h)) * S_SEQ + s) * HD_DIM + d;
                    else
                        idx = (((size_t)(b * NHEADS + h)) * HD_DIM + d) * S_SEQ + s;
                    ((__bf16*)Cout)[idx] = (__bf16)val;
                }
            }
        }
    }
}

// ---------------- Flash attention (block-cooperative) ----------------
// Block = 8 waves, all on the same (b,h). Wave w owns query rows q0..q0+15.
// Each iteration stages the shared 32x128 K tile and 128x32 V^T tile into LDS
// (async global->LDS when available), then all waves consume them.
__global__ __launch_bounds__(256) void attn_flash(const __bf16* __restrict__ Qh,
                                                  const __bf16* __restrict__ Kh,
                                                  const __bf16* __restrict__ Vt,
                                                  __bf16* __restrict__ Ctx) {
    __shared__ __bf16 shK[32 * 128];       // K tile  [j_rel][d]      8 KB
    __shared__ __bf16 shV[128 * 32];       // V tile  [d][j_rel]      8 KB
    __shared__ __bf16 shQ[8][16 * 128];    // per-wave Q tile        32 KB
    __shared__ __bf16 shP[8][16 * 32];     // per-wave P staging      8 KB

    const int tid    = threadIdx.x;
    const int lane   = tid & 31;
    const int wave   = tid >> 5;
    const int col    = lane & 15;
    const int aHalfB = (lane & 16) ? 16 : 0;
    const int bHalfB = (lane & 16) ? 32 : 0;
    const int mhalf  = (lane & 16) ? 8 : 0;

    const int bh   = blockIdx.x >> 4;      // 16 blocks per (b,h)
    const int qblk = blockIdx.x & 15;
    const int q0   = qblk * 128 + wave * 16;
    const int b    = bh >> 3;
    const int h    = bh & 7;

    const float sc_log2 = 0.08838834764831845f * 1.4426950408889634f; // 1/sqrt(HD)*log2(e)

    const __bf16* Qbh = Qh + (size_t)bh * S_SEQ * HD_DIM;
    const __bf16* Kbh = Kh + (size_t)bh * S_SEQ * HD_DIM;
    const __bf16* Vbh = Vt + (size_t)bh * HD_DIM * S_SEQ;

    // ---- stage this wave's Q tile (16x128) into LDS: 8x16B per lane
#pragma unroll
    for (int i = 0; i < 8; ++i) {
        int cq  = lane + 32 * i;           // 16B chunk id, 256 total
        int row = cq >> 4, c8 = cq & 15;
        async_copy16(Qbh + (size_t)(q0 + row) * HD_DIM + c8 * 8, &shQ[wave][cq * 8]);
    }

    v8f o[8];
#pragma unroll
    for (int t = 0; t < 8; ++t) o[t] = zero8();
    float mrun[8], lrun[8];
#pragma unroll
    for (int r = 0; r < 8; ++r) { mrun[r] = -1e30f; lrun[r] = 0.f; }

    for (int j0 = 0; j0 < S_SEQ; j0 += 32) {
        __syncthreads();                   // previous tile fully consumed
        // ---- stage K (32x128) and V (128x32): 512 x16B chunks each, 2 per thread
#pragma unroll
        for (int i = 0; i < 2; ++i) {
            int c = tid + 256 * i;
            async_copy16(Kbh + (size_t)(j0 + (c >> 4)) * HD_DIM + (c & 15) * 8,
                         &shK[c * 8]);
            async_copy16(Vbh + (size_t)(c >> 2) * S_SEQ + j0 + (c & 3) * 8,
                         &shV[c * 8]);
        }
        async_join();
        __syncthreads();                   // staged data visible to all waves

        // ---- scores: S = Q K^T over this 32-wide key tile
        v8f s0 = zero8(), s1 = zero8();
#pragma unroll
        for (int c = 0; c < 4; ++c) {
            FragBF qa, kb;
            const char* qrow = (const char*)&shQ[wave][col * 128] + 64 * c;
            qa.q[0] = *(const uint4*)(qrow + aHalfB);
            qa.q[1] = *(const uint4*)(qrow + 32 + aHalfB);
            const char* krow0 = (const char*)&shK[col * 128] + 64 * c;
            kb.q[0] = *(const uint4*)(krow0 + aHalfB);
            kb.q[1] = *(const uint4*)(krow0 + 32 + aHalfB);
            s0 = wmma_bf16(qa, kb, s0);
            const char* krow1 = krow0 + 16 * 256;   // +16 rows of 256B
            kb.q[0] = *(const uint4*)(krow1 + aHalfB);
            kb.q[1] = *(const uint4*)(krow1 + 32 + aHalfB);
            s1 = wmma_bf16(qa, kb, s1);
        }

        // ---- online softmax per row (reduce across 16-lane half), fold rescale
#pragma unroll
        for (int r = 0; r < 8; ++r) {
            float mx = fmaxf(s0[r], s1[r]);
#pragma unroll
            for (int off = 1; off < 16; off <<= 1)
                mx = fmaxf(mx, __shfl_xor(mx, off, 32));
            float mnew = fmaxf(mrun[r], mx);
            float a  = exp2f((mrun[r] - mnew) * sc_log2);
            float p0 = exp2f((s0[r] - mnew) * sc_log2);
            float p1 = exp2f((s1[r] - mnew) * sc_log2);
            s0[r] = p0;
            s1[r] = p1;
            float ps = p0 + p1;
#pragma unroll
            for (int off = 1; off < 16; off <<= 1)
                ps += __shfl_xor(ps, off, 32);
            lrun[r] = lrun[r] * a + ps;
            mrun[r] = mnew;
#pragma unroll
            for (int t = 0; t < 8; ++t) o[t][r] *= a;
        }

        // ---- stage P (16x32 bf16) into this wave's LDS slab
        {
            __bf16* pw = &shP[wave][0];
#pragma unroll
            for (int r = 0; r < 8; ++r) {
                pw[(r + mhalf) * 32 + col]      = (__bf16)s0[r];
                pw[(r + mhalf) * 32 + 16 + col] = (__bf16)s1[r];
            }
        }
        asm volatile("s_wait_dscnt 0" ::: "memory");

        FragBF pa;
        {
            const char* prow = (const char*)&shP[wave][col * 32];
            pa.q[0] = *(const uint4*)(prow + aHalfB);
            pa.q[1] = *(const uint4*)(prow + 32 + aHalfB);
        }

        // ---- O += P V
#pragma unroll
        for (int t = 0; t < 8; ++t) {
            FragBF vb;
            const char* vrow = (const char*)&shV[(t * 16 + col) * 32];
            vb.q[0] = *(const uint4*)(vrow + aHalfB);
            vb.q[1] = *(const uint4*)(vrow + 32 + aHalfB);
            o[t] = wmma_bf16(pa, vb, o[t]);
        }
    }

    // ---- normalize and store context to [B,S,H]
#pragma unroll
    for (int r = 0; r < 8; ++r) {
        float inv = 1.f / lrun[r];
        int s = q0 + r + mhalf;
        __bf16* dst = Ctx + ((size_t)b * S_SEQ + s) * H_DIM + h * HD_DIM + col;
#pragma unroll
        for (int t = 0; t < 8; ++t)
            dst[t * 16] = (__bf16)(o[t][r] * inv);
    }
}

// ---------------- host launcher ----------------
extern "C" void kernel_launch(void* const* d_in, const int* in_sizes, int n_in,
                              void* d_out, int out_size, void* d_ws, size_t ws_size,
                              hipStream_t stream) {
    const float* x  = (const float*)d_in[0];
    const float* Wq = (const float*)d_in[1];
    const float* Wk = (const float*)d_in[2];
    const float* Wv = (const float*)d_in[3];
    const float* Wo = (const float*)d_in[4];
    float* out = (float*)d_out;

    const int M = B_SZ * S_SEQ;     // 8192
    const int N = H_DIM;            // 1024
    const int K = H_DIM;            // 1024
    const size_t nX = (size_t)M * K;
    const size_t nW = (size_t)H_DIM * H_DIM;
    const size_t nQ = (size_t)B_SZ * NHEADS * S_SEQ * HD_DIM;

    char* ws = (char*)d_ws;
    size_t off = 0;
    __bf16* Xbf  = (__bf16*)(ws + off); off += nX * 2;
    __bf16* Wqb  = (__bf16*)(ws + off); off += nW * 2;
    __bf16* Wkb  = (__bf16*)(ws + off); off += nW * 2;
    __bf16* Wvb  = (__bf16*)(ws + off); off += nW * 2;
    __bf16* Wob  = (__bf16*)(ws + off); off += nW * 2;
    __bf16* Qh   = (__bf16*)(ws + off); off += nQ * 2;
    __bf16* Kh   = (__bf16*)(ws + off); off += nQ * 2;
    __bf16* Vt   = (__bf16*)(ws + off); off += nQ * 2;
    __bf16* Ctx  = (__bf16*)(ws + off); off += nX * 2;

    // 1) fp32 -> bf16 staging
    {
        int n4 = (int)(nX / 4);
        cvt_f32_to_bf16_v4<<<n4 / 256, 256, 0, stream>>>(x, Xbf, n4);
        int w4 = (int)(nW / 4);
        cvt_f32_to_bf16_v4<<<w4 / 256, 256, 0, stream>>>(Wq, Wqb, w4);
        cvt_f32_to_bf16_v4<<<w4 / 256, 256, 0, stream>>>(Wk, Wkb, w4);
        cvt_f32_to_bf16_v4<<<w4 / 256, 256, 0, stream>>>(Wv, Wvb, w4);
        cvt_f32_to_bf16_v4<<<w4 / 256, 256, 0, stream>>>(Wo, Wob, w4);
    }

    // 2) projections: Q, K natural head layout; V transposed per head
    dim3 ggrid(M / 32, N / 512);
    gemm_xwt<0><<<ggrid, 256, 0, stream>>>(Xbf, Wqb, (void*)Qh, M, N, K);
    gemm_xwt<0><<<ggrid, 256, 0, stream>>>(Xbf, Wkb, (void*)Kh, M, N, K);
    gemm_xwt<1><<<ggrid, 256, 0, stream>>>(Xbf, Wvb, (void*)Vt, M, N, K);

    // 3) flash attention: 32 (b,h) x 16 q-blocks = 512 blocks of 8 waves
    attn_flash<<<B_SZ * NHEADS * (S_SEQ / 128), 256, 0, stream>>>(Qh, Kh, Vt, Ctx);

    // 4) output projection -> fp32 d_out
    gemm_xwt<2><<<ggrid, 256, 0, stream>>>(Ctx, Wob, (void*)out, M, N, K);
}
